// appnp_66228395705230
// MI455X (gfx1250) — matrix operand order repaired and checked by reading, not compiled, not run on hardware
//
#include <hip/hip_runtime.h>

// ---------------- problem constants ----------------
constexpr int Nn   = 100000;   // nodes
constexpr int FIN  = 512;      // input features
constexpr int HID  = 256;      // hidden
constexpr int Cc   = 40;       // classes
constexpr int CP   = 48;       // classes padded to 3x16 WMMA tiles
constexpr int Ec   = 1600000;  // edges
constexpr int Kp   = 10;       // propagation steps

// ---------------- WMMA types ----------------
typedef __attribute__((ext_vector_type(16))) __bf16 v16bf;
typedef __attribute__((ext_vector_type(8)))  float  v8f;

union ABfrag { unsigned int u[8]; v16bf v; };

__device__ __forceinline__ unsigned short f32_to_bf16(float f) {
  unsigned int u = __float_as_uint(f);
  u = u + 0x7FFFu + ((u >> 16) & 1u);   // round-to-nearest-even
  return (unsigned short)(u >> 16);
}
__device__ __forceinline__ unsigned int pk_bf16(float lo, float hi) {
  unsigned int ul = __float_as_uint(lo);
  unsigned int uh = __float_as_uint(hi);
  ul = (ul + 0x7FFFu + ((ul >> 16) & 1u)) >> 16;
  uh = (uh + 0x7FFFu + ((uh >> 16) & 1u)) & 0xFFFF0000u;
  return ul | uh;
}
__device__ __forceinline__ void load_bfrag(ABfrag& b, const unsigned short* bp) {
  const uint4 b0 = *(const uint4*)(bp);
  const uint4 b1 = *(const uint4*)(bp + 16);
  b.u[0] = b0.x; b.u[1] = b0.y; b.u[2] = b0.z; b.u[3] = b0.w;
  b.u[4] = b1.x; b.u[5] = b1.y; b.u[6] = b1.z; b.u[7] = b1.w;
}

// ---------------- weight transpose/convert kernels ----------------
// W1 [FIN,HID] f32 row-major  ->  W1t [HID][FIN] bf16 (col-major of original)
__global__ void w1t_prep(const float* __restrict__ W1, unsigned short* __restrict__ W1t) {
  int tid = blockIdx.x * blockDim.x + threadIdx.x;
  if (tid >= HID * FIN) return;
  int c = tid / FIN, k = tid % FIN;
  W1t[(size_t)c * FIN + k] = f32_to_bf16(W1[(size_t)k * HID + c]);
}
// W2 [HID,Cc] f32 -> W2t [CP][HID] bf16, zero-padded cols 40..47
__global__ void w2t_prep(const float* __restrict__ W2, unsigned short* __restrict__ W2t) {
  int tid = blockIdx.x * blockDim.x + threadIdx.x;
  if (tid >= CP * HID) return;
  int c = tid / HID, k = tid % HID;
  W2t[(size_t)c * HID + k] = (c < Cc) ? f32_to_bf16(W2[(size_t)k * Cc + c]) : (unsigned short)0;
}

// ---------------- degree / edge-weight kernels ----------------
__global__ void deg_init(float* __restrict__ deg) {
  int i = blockIdx.x * blockDim.x + threadIdx.x;
  if (i < Nn) deg[i] = 1.0f;                    // self-loop
}
__global__ void deg_scatter(const int* __restrict__ dst, float* __restrict__ deg) {
  int e = blockIdx.x * blockDim.x + threadIdx.x;
  if (e < Ec) unsafeAtomicAdd(&deg[dst[e]], 1.0f);
}
__global__ void dinv_k(const float* __restrict__ deg, float* __restrict__ dinv) {
  int i = blockIdx.x * blockDim.x + threadIdx.x;
  if (i < Nn) dinv[i] = rsqrtf(deg[i]);         // deg >= 1 always
}
// w[e] = (1-alpha) * dinv[src] * dinv[dst]
__global__ void wgt_k(const int* __restrict__ ei, const float* __restrict__ dinv,
                      float* __restrict__ w) {
  int e = blockIdx.x * blockDim.x + threadIdx.x;
  if (e < Ec) w[e] = 0.9f * dinv[ei[e]] * dinv[ei[Ec + e]];
}

// ---------------- GEMM1: h = relu(x @ W1 + b1), WMMA bf16 ----------------
// block: 128 threads (4 waves), tile = 16 rows x 256 cols; wave w -> cols [64w,64w+64)
// Double-buffered LDS A tile (1 barrier/iter) + all-B-loads-then-WMMA pipelining.
__global__ __launch_bounds__(128) void gemm1_k(const float* __restrict__ x,
                                               const unsigned short* __restrict__ W1t,
                                               const float* __restrict__ bias1,
                                               float* __restrict__ hout) {
  __shared__ __align__(16) unsigned short lA[2][16][32];   // 2x 16x32 bf16 A tiles
  const int tid  = threadIdx.x;
  const int lane = tid & 31;
  const int wave = tid >> 5;
  const int rtile = blockIdx.x * 16;
  const int row16 = lane & 15;
  const int akb   = (lane >> 4) * 8;   // A: lanes0-15 K{0..7,16..23}; lanes16-31 K{8..15,24..31}
  const int c0    = wave * 64;
  const int cr    = tid >> 3;          // coop-load row
  const int ckq   = (tid & 7) * 4;     // coop-load K quad

  v8f acc[4] = {};

  // prologue: stage K-block 0 into buffer 0
  {
    const float4 f = *(const float4*)(x + (size_t)(rtile + cr) * FIN + ckq);
    uint2 p; p.x = pk_bf16(f.x, f.y); p.y = pk_bf16(f.z, f.w);
    *(uint2*)(&lA[0][cr][ckq]) = p;
  }

  constexpr int KIT = FIN / 32;
  for (int i = 0; i < KIT; ++i) {
    __syncthreads();                         // buf[i&1] ready; buf[(i+1)&1] free
    if (i + 1 < KIT) {                       // prefetch next A tile into other buffer
      const float4 f = *(const float4*)(x + (size_t)(rtile + cr) * FIN + (i + 1) * 32 + ckq);
      uint2 p; p.x = pk_bf16(f.x, f.y); p.y = pk_bf16(f.z, f.w);
      *(uint2*)(&lA[(i + 1) & 1][cr][ckq]) = p;
    }

    ABfrag a;
    {
      const uint4 a0 = *(const uint4*)(&lA[i & 1][row16][akb]);
      const uint4 a1 = *(const uint4*)(&lA[i & 1][row16][akb + 16]);
      a.u[0] = a0.x; a.u[1] = a0.y; a.u[2] = a0.z; a.u[3] = a0.w;
      a.u[4] = a1.x; a.u[5] = a1.y; a.u[6] = a1.z; a.u[7] = a1.w;
    }

    const int kb2 = i * 32 + ((lane >> 4) ? 8 : 0);  // B: lanes0-15 K{0..7,16..23}; lanes16-31 K{8..15,24..31}
    ABfrag b[4];
#pragma unroll
    for (int t = 0; t < 4; ++t) {
      const int col = c0 + t * 16 + (lane & 15);
      load_bfrag(b[t], W1t + (size_t)col * FIN + kb2);
    }
#pragma unroll
    for (int t = 0; t < 4; ++t)
      acc[t] = __builtin_amdgcn_wmma_f32_16x16x32_bf16(false, a.v, false, b[t].v,
                                                       (short)0, acc[t], false, false);
  }

  // epilogue: bias + relu, C layout: (M = r + 8*lane[4], N = lane[3:0])
  const int rbase = rtile + ((lane >> 4) << 3);
#pragma unroll
  for (int t = 0; t < 4; ++t) {
    const int col  = c0 + t * 16 + (lane & 15);
    const float bv = bias1[col];
#pragma unroll
    for (int r = 0; r < 8; ++r) {
      float v = acc[t][r] + bv;
      hout[(size_t)(rbase + r) * HID + col] = v > 0.0f ? v : 0.0f;
    }
  }
}

// ---------------- GEMM2: h2 = relu(h @ W2 + b2), WMMA bf16 ----------------
// block: 128 threads (4 waves); wave w -> rows [64*blk + 16w, +16), cols 0..47 (3 tiles)
__global__ __launch_bounds__(128) void gemm2_k(const float* __restrict__ h,
                                               const unsigned short* __restrict__ W2t,
                                               const float* __restrict__ bias2,
                                               float* __restrict__ h2) {
  const int lane  = threadIdx.x & 31;
  const int wave  = threadIdx.x >> 5;
  const int rtile = blockIdx.x * 64 + wave * 16;
  if (rtile >= Nn) return;                 // Nn % 16 == 0, so whole-wave guard is exact
  const int row = rtile + (lane & 15);
  const int akb = (lane >> 4) * 8;

  v8f acc[3] = {};

  for (int kb = 0; kb < HID; kb += 32) {
    const float* ap = h + (size_t)row * HID + kb + akb;
    const float4 f0 = *(const float4*)(ap);
    const float4 f1 = *(const float4*)(ap + 4);
    const float4 f2 = *(const float4*)(ap + 16);
    const float4 f3 = *(const float4*)(ap + 20);
    ABfrag a;
    a.u[0] = pk_bf16(f0.x, f0.y); a.u[1] = pk_bf16(f0.z, f0.w);
    a.u[2] = pk_bf16(f1.x, f1.y); a.u[3] = pk_bf16(f1.z, f1.w);
    a.u[4] = pk_bf16(f2.x, f2.y); a.u[5] = pk_bf16(f2.z, f2.w);
    a.u[6] = pk_bf16(f3.x, f3.y); a.u[7] = pk_bf16(f3.z, f3.w);

    const int kb2 = kb + ((lane >> 4) ? 8 : 0);
    ABfrag b[3];
#pragma unroll
    for (int t = 0; t < 3; ++t) {
      const int col = t * 16 + (lane & 15);
      load_bfrag(b[t], W2t + (size_t)col * HID + kb2);
    }
#pragma unroll
    for (int t = 0; t < 3; ++t)
      acc[t] = __builtin_amdgcn_wmma_f32_16x16x32_bf16(false, a.v, false, b[t].v,
                                                       (short)0, acc[t], false, false);
  }

  const int rbase = rtile + ((lane >> 4) << 3);
#pragma unroll
  for (int t = 0; t < 3; ++t) {
    const int col = t * 16 + (lane & 15);
    if (col < Cc) {
      const float bv = bias2[col];
#pragma unroll
      for (int r = 0; r < 8; ++r) {
        float v = acc[t][r] + bv;
        h2[(size_t)(rbase + r) * Cc + col] = v > 0.0f ? v : 0.0f;
      }
    }
  }
}

// ---------------- propagation ----------------
// znext = alpha*h2 + (1-alpha)*dinv[n]^2 * zcur   (teleport + self-loop term), float4-vectorized
__global__ void prop_init(const float* __restrict__ h2, const float* __restrict__ zcur,
                          const float* __restrict__ dinv, float* __restrict__ znext) {
  int tid = blockIdx.x * blockDim.x + threadIdx.x;
  if (tid >= Nn * (Cc / 4)) return;
  int n = tid / (Cc / 4);
  float di = dinv[n];
  float s = 0.9f * di * di;
  const float4 hv = ((const float4*)h2)[tid];
  const float4 zv = ((const float4*)zcur)[tid];
  float4 o;
  o.x = 0.1f * hv.x + s * zv.x;
  o.y = 0.1f * hv.y + s * zv.y;
  o.z = 0.1f * hv.z + s * zv.z;
  o.w = 0.1f * hv.w + s * zv.w;
  ((float4*)znext)[tid] = o;
}

// edge scatter: 5 threads per edge, 8 classes each; z buffers stay L2-resident (32 MB << 192 MB)
__global__ void prop_scatter(const int* __restrict__ ei, const float* __restrict__ w,
                             const float* __restrict__ zcur, float* __restrict__ znext) {
  int tid = blockIdx.x * blockDim.x + threadIdx.x;
  if (tid >= Ec * 5) return;
  int e = tid / 5;
  int g = (tid - e * 5) * 8;
  const int   s  = ei[e];
  const int   d  = ei[Ec + e];
  const float ww = w[e];
  const float4 z0 = *(const float4*)(zcur + (size_t)s * Cc + g);
  const float4 z1 = *(const float4*)(zcur + (size_t)s * Cc + g + 4);
  float* o = znext + (size_t)d * Cc + g;
  unsafeAtomicAdd(o + 0, ww * z0.x);
  unsafeAtomicAdd(o + 1, ww * z0.y);
  unsafeAtomicAdd(o + 2, ww * z0.z);
  unsafeAtomicAdd(o + 3, ww * z0.w);
  unsafeAtomicAdd(o + 4, ww * z1.x);
  unsafeAtomicAdd(o + 5, ww * z1.y);
  unsafeAtomicAdd(o + 6, ww * z1.z);
  unsafeAtomicAdd(o + 7, ww * z1.w);
}

// ---------------- host launcher ----------------
extern "C" void kernel_launch(void* const* d_in, const int* in_sizes, int n_in,
                              void* d_out, int out_size, void* d_ws, size_t ws_size,
                              hipStream_t stream) {
  const float* x   = (const float*)d_in[0];
  const int*   ei  = (const int*)  d_in[1];
  const float* W1  = (const float*)d_in[2];
  const float* b1  = (const float*)d_in[3];
  const float* W2  = (const float*)d_in[4];
  const float* b2  = (const float*)d_in[5];
  float* out = (float*)d_out;

  char* ws = (char*)d_ws;
  size_t off = 0;
  auto alloc = [&](size_t bytes) -> void* {
    void* p = ws + off;
    off = (off + bytes + 255) & ~(size_t)255;
    return p;
  };
  float*          h1   = (float*)         alloc((size_t)Nn * HID * 4);
  float*          h2   = (float*)         alloc((size_t)Nn * Cc * 4);
  float*          zA   = (float*)         alloc((size_t)Nn * Cc * 4);
  float*          zB   = (float*)         alloc((size_t)Nn * Cc * 4);
  float*          deg  = (float*)         alloc((size_t)Nn * 4);
  float*          dinv = (float*)         alloc((size_t)Nn * 4);
  float*          w    = (float*)         alloc((size_t)Ec * 4);
  unsigned short* W1t  = (unsigned short*)alloc((size_t)HID * FIN * 2);
  unsigned short* W2t  = (unsigned short*)alloc((size_t)CP * HID * 2);

  // weights -> bf16 transposed
  w1t_prep<<<(HID * FIN + 255) / 256, 256, 0, stream>>>(W1, W1t);
  w2t_prep<<<(CP * HID + 255) / 256, 256, 0, stream>>>(W2, W2t);

  // degrees / normalization / edge weights
  deg_init   <<<(Nn + 255) / 256, 256, 0, stream>>>(deg);
  deg_scatter<<<(Ec + 255) / 256, 256, 0, stream>>>(ei + Ec, deg);
  dinv_k     <<<(Nn + 255) / 256, 256, 0, stream>>>(deg, dinv);
  wgt_k      <<<(Ec + 255) / 256, 256, 0, stream>>>(ei, dinv, w);

  // MLP front-end (WMMA bf16)
  gemm1_k<<<Nn / 16, 128, 0, stream>>>(x, W1t, b1, h1);
  gemm2_k<<<(Nn + 63) / 64, 128, 0, stream>>>(h1, W2t, b2, h2);

  // K-step APPNP power iteration; final step writes d_out
  const float* zc = h2;
  float* pp[2] = { zA, zB };
  for (int it = 0; it < Kp; ++it) {
    float* zn = (it == Kp - 1) ? out : pp[it & 1];
    prop_init   <<<(Nn * (Cc / 4) + 255) / 256, 256, 0, stream>>>(h2, zc, dinv, zn);
    prop_scatter<<<(Ec * 5 + 255) / 256, 256, 0, stream>>>(ei, w, zc, zn);
    zc = zn;
  }
}
